// MultiheadAttention_8546984919258
// MI455X (gfx1250) — compile-verified, matrix operand order checked by reading, not run on hardware
//
#include <hip/hip_runtime.h>
#include <hip/hip_bf16.h>

// MI455X / gfx1250 implementation.
// Pipeline: 3x WMMA GEMM projections -> per-token stats -> V transpose ->
// fused entropy-score attention (flash-style; scores provably tiny so no
// running max needed) with TDM-staged V tiles and WMMA P@V ->
// segmented cumsum -> WMMA FC GEMM.
//
// Workspace layout (bytes), total ~129 MB:
//   Qf32   @ 0        (16384*512*4 = 32 MB)
//   Kf32   @ 32 MB    (32 MB)
//   AO     @ 64 MB    (attn output / cumsum in-place, 32 MB)
//   Vh     @ 96 MB    (f16 row-major [B,L,D], 16 MB)
//   VT     @ 112 MB   (f16 transposed [B,D,L], 16 MB)
//   stats  @ 128 MB   (8 arrays * 16384 f32 = 512 KB)
//   aux    @ 128.5 MB (8*8*512 f32 = 128 KB)

typedef __attribute__((ext_vector_type(16))) _Float16     v16h;
typedef __attribute__((ext_vector_type(8)))  float        v8f;
typedef __attribute__((ext_vector_type(4)))  unsigned int u32x4;
typedef __attribute__((ext_vector_type(4)))  int          i32x4;
typedef __attribute__((ext_vector_type(8)))  int          i32x8;

#if defined(__has_builtin)
#  if __has_builtin(__builtin_amdgcn_tensor_load_to_lds) && \
      __has_builtin(__builtin_amdgcn_s_wait_tensorcnt)
#    define HAVE_TDM 1
#  endif
#endif
#ifndef HAVE_TDM
#  define HAVE_TDM 0
#endif

#define BM 128
#define BN 64
#define BK 32

// ---------------------------------------------------------------------------
// GEMM: out[m,n] = sum_k X[m,k] * W[n,k] + bias[n]   (einsum 'bld,ed->ble')
// Both A and B are K-contiguous in memory -> identical fragment load pattern.
// Block tile 128x64, 8 waves (4x2), each wave 32x32 via 2x2 WMMA 16x16x32.
// ---------------------------------------------------------------------------
template <bool WRITE_F16>
__global__ __launch_bounds__(256) void gemm_xwT(
    const float* __restrict__ X, const float* __restrict__ W,
    const float* __restrict__ bias, void* __restrict__ outP,
    int M, int N, int K)
{
    __shared__ __align__(32) _Float16 As[BM * BK];  // 8 KB
    __shared__ __align__(32) _Float16 Bs[BN * BK];  // 4 KB

    const int tid  = threadIdx.x;
    const int m0   = blockIdx.y * BM;
    const int n0   = blockIdx.x * BN;
    const int w    = tid >> 5;             // wave 0..7
    const int wm   = (w >> 1) * 32;        // wave M offset: 0,32,64,96
    const int wn   = (w & 1) * 32;         // wave N offset: 0,32
    const int lane = tid & 31;
    const int lr   = lane & 15;            // row/col within 16
    const int khv  = lane >> 4;            // K-half select

    v8f acc[2][2] = {};

    for (int kk = 0; kk < K; kk += BK) {
        // Stage A tile: 128 rows x 32 K, f32 -> f16.  Thread: 16 contiguous.
        {
            const int row  = tid >> 1;
            const int half = (tid & 1) * 16;
            const float* src = X + (size_t)(m0 + row) * K + kk + half;
            __builtin_prefetch(src + BK, 0, 1);   // global_prefetch next K tile
            _Float16* dst = &As[row * BK + half];
            #pragma unroll
            for (int i = 0; i < 16; ++i) dst[i] = (_Float16)src[i];
        }
        // Stage B tile: 64 rows x 32 K.  Thread: 8 contiguous.
        {
            const int row = tid >> 2;
            const int q   = (tid & 3) * 8;
            const float* src = W + (size_t)(n0 + row) * K + kk + q;
            _Float16* dst = &Bs[row * BK + q];
            #pragma unroll
            for (int i = 0; i < 8; ++i) dst[i] = (_Float16)src[i];
        }
        __syncthreads();

        v16h a[2], b[2];
        #pragma unroll
        for (int mi = 0; mi < 2; ++mi)
            a[mi] = *(const v16h*)&As[(wm + mi * 16 + lr) * BK + khv * 16];
        #pragma unroll
        for (int ni = 0; ni < 2; ++ni)
            b[ni] = *(const v16h*)&Bs[(wn + ni * 16 + lr) * BK + khv * 16];

        #pragma unroll
        for (int mi = 0; mi < 2; ++mi)
            #pragma unroll
            for (int ni = 0; ni < 2; ++ni)
                acc[mi][ni] = __builtin_amdgcn_wmma_f32_16x16x32_f16(
                    false, a[mi], false, b[ni], (short)0, acc[mi][ni],
                    false, false);
        __syncthreads();
    }

    // Epilogue.  C layout: element (vgpr v, lane): M = v + 8*(lane>>4), N = lane&15.
    const int crow = (lane >> 4) * 8;
    const int ccol = lane & 15;
    #pragma unroll
    for (int mi = 0; mi < 2; ++mi) {
        #pragma unroll
        for (int ni = 0; ni < 2; ++ni) {
            const int gn = n0 + wn + ni * 16 + ccol;
            const float bv = bias[gn];
            #pragma unroll
            for (int v = 0; v < 8; ++v) {
                const int gm = m0 + wm + mi * 16 + crow + v;
                const float val = acc[mi][ni][v] + bv;
                if constexpr (WRITE_F16)
                    ((_Float16*)outP)[(size_t)gm * N + gn] = (_Float16)val;
                else
                    ((float*)outP)[(size_t)gm * N + gn] = val;
            }
        }
    }
}

// ---------------------------------------------------------------------------
// Per-token shifted-exp stats over D=512: m = max, Z = sum exp(x-m),
// S = sum exp(x-m)*x, h = (S/Z - (m + log Z)) / D.  One wave per token.
// ---------------------------------------------------------------------------
__global__ __launch_bounds__(256) void token_stats(
    const float* __restrict__ X,
    float* __restrict__ om, float* __restrict__ oz,
    float* __restrict__ os, float* __restrict__ oh, int D)
{
    const int row  = blockIdx.x * 8 + (threadIdx.x >> 5);
    const int lane = threadIdx.x & 31;
    const float* src = X + (size_t)row * D + lane * 16;

    float v[16];
    #pragma unroll
    for (int i = 0; i < 16; ++i) v[i] = src[i];

    float m = v[0];
    #pragma unroll
    for (int i = 1; i < 16; ++i) m = fmaxf(m, v[i]);
    #pragma unroll
    for (int o = 16; o >= 1; o >>= 1) m = fmaxf(m, __shfl_xor(m, o, 32));

    float z = 0.0f, s = 0.0f;
    #pragma unroll
    for (int i = 0; i < 16; ++i) {
        const float e = __expf(v[i] - m);
        z += e;
        s += e * v[i];
    }
    #pragma unroll
    for (int o = 16; o >= 1; o >>= 1) {
        z += __shfl_xor(z, o, 32);
        s += __shfl_xor(s, o, 32);
    }
    if (lane == 0) {
        om[row] = m;
        oz[row] = z;
        os[row] = s;
        oh[row] = (s / z - (m + __logf(z))) / (float)D;
    }
}

// ---------------------------------------------------------------------------
// Transpose V:  [B, L, D] f16 -> [B, D, L] f16, 64x64 LDS tiles.
// Makes the attention-side V tile a pure 2D strided-row copy (TDM-friendly)
// and makes WMMA B-fragments contiguous.
// ---------------------------------------------------------------------------
__global__ __launch_bounds__(256) void transpose_v(
    const _Float16* __restrict__ src, _Float16* __restrict__ dst, int L, int D)
{
    __shared__ _Float16 tile[64][72];   // pad 8 to dodge bank conflicts
    const int b  = blockIdx.z;
    const int j0 = blockIdx.x * 64;
    const int d0 = blockIdx.y * 64;
    const int t  = threadIdx.x;
    const int r  = t >> 2;              // 0..63
    const int cc = (t & 3) * 16;        // 0,16,32,48

    const _Float16* s = src + ((size_t)b * L + j0 + r) * D + d0 + cc;
    #pragma unroll
    for (int u = 0; u < 16; ++u) tile[r][cc + u] = s[u];
    __syncthreads();

    _Float16* o = dst + ((size_t)b * D + d0 + r) * L + j0 + cc;
    #pragma unroll
    for (int u = 0; u < 16; ++u) o[u] = tile[cc + u][r];
}

// ---------------------------------------------------------------------------
// Fused attention: scores[i,j] from per-token scalars only.
// |score| <= ~3*log(1024)/1024/8 -> tiny, so exp() is safe without a running
// max; softmax = exp(s)/rowsum directly.
// Phase A: rowsum_i = sum_j exp(s_ij)   (pure VALU exp/log from stats).
// Phase B: loop j tiles of 32: build P f16 in LDS, DMA V^T tile into LDS via
//          the Tensor Data Mover (one descriptor per wave, 64 d-rows x 32 j),
//          acc += P @ V via WMMA (each wave owns 64 output channels).
// Grid: (L/32, B), block 256 (8 waves).
// ---------------------------------------------------------------------------
__global__ __launch_bounds__(256) void attn_pv(
    const float* __restrict__ qm, const float* __restrict__ qz,
    const float* __restrict__ qs, const float* __restrict__ qh,
    const float* __restrict__ km, const float* __restrict__ kz,
    const float* __restrict__ ks, const float* __restrict__ khs,
    const _Float16* __restrict__ VT,   // [B, D, L] f16
    float* __restrict__ out, int L, int D)
{
    __shared__ __align__(32) _Float16 Pls[32 * 32];   // P tile  [i][j], 2 KB
    __shared__ __align__(64) _Float16 Vt[512 * 32];   // V^T     [d][j], 32 KB
    __shared__ float kchunk[4 * 256];                 // K-stat staging
    __shared__ float partial[32 * 8];
    __shared__ float rinvLds[32];

    const int b    = blockIdx.y;
    const int i0   = blockIdx.x * 32;
    const int tid  = threadIdx.x;
    const int ti   = tid >> 3;   // i row 0..31
    const int tj   = tid & 7;
    const int w    = tid >> 5;
    const int lane = tid & 31;
    const int lr   = lane & 15;
    const int khv  = lane >> 4;
    const int d0   = w * 64;     // each wave owns 64 channels

    const size_t tokBase = (size_t)b * L;
    const float miq = qm[tokBase + i0 + ti];
    const float ziq = qz[tokBase + i0 + ti];
    const float siq = qs[tokBase + i0 + ti];
    const float hiq = qh[tokBase + i0 + ti];
    const float invScale = 0.125f;          // 1/sqrt(512/8)
    const float inv2D    = 1.0f / 1024.0f;  // 1/(2*D1)

    // ---- Phase A: row sums of exp(scores) ----
    float rsum = 0.0f;
    for (int jc = 0; jc < L; jc += 256) {
        kchunk[      tid] = km [tokBase + jc + tid];
        kchunk[256 + tid] = kz [tokBase + jc + tid];
        kchunk[512 + tid] = ks [tokBase + jc + tid];
        kchunk[768 + tid] = khs[tokBase + jc + tid];
        __syncthreads();
        #pragma unroll 4
        for (int n = 0; n < 32; ++n) {
            const int j = tj + n * 8;
            const float mk = kchunk[j],       zk = kchunk[256 + j];
            const float sk = kchunk[512 + j], hk = kchunk[768 + j];
            const float Mx = fmaxf(miq, mk);
            const float wa = __expf(miq - Mx), wb = __expf(mk - Mx);
            const float Zp = wa * ziq + wb * zk;
            const float Sp = wa * siq + wb * sk;
            const float hqk = (Sp / Zp - (Mx + __logf(Zp))) * inv2D;
            rsum += __expf(-(hiq + hk - hqk) * invScale);
        }
        __syncthreads();
    }
    partial[ti * 8 + tj] = rsum;
    __syncthreads();
    if (tid < 32) {
        float s = 0.0f;
        #pragma unroll
        for (int q = 0; q < 8; ++q) s += partial[tid * 8 + q];
        rinvLds[tid] = 1.0f / s;
    }
    __syncthreads();

    // ---- Phase B: acc += P @ V ----
    v8f acc[2][4] = {};

    for (int jt = 0; jt < L; jt += 32) {
        // P tile: thread computes 4 contiguous j for its row ti.
        {
            const int jj0 = tj * 4;
            #pragma unroll
            for (int u = 0; u < 4; ++u) {
                const int j = jt + jj0 + u;
                const float mk = km[tokBase + j],  zk = kz[tokBase + j];
                const float sk = ks[tokBase + j],  hk = khs[tokBase + j];
                const float Mx = fmaxf(miq, mk);
                const float wa = __expf(miq - Mx), wb = __expf(mk - Mx);
                const float Zp = wa * ziq + wb * zk;
                const float Sp = wa * siq + wb * sk;
                const float hqk = (Sp / Zp - (Mx + __logf(Zp))) * inv2D;
                const float p = __expf(-(hiq + hk - hqk) * invScale);
                Pls[ti * 32 + jj0 + u] = (_Float16)p;
            }
        }

        // Stage V^T tile [512 d][32 j] into LDS.  TDM path: each wave DMAs
        // its own 64 d-rows x 32 j sub-tile (rows contiguous 64 B in LDS).
#if HAVE_TDM
        {
            const unsigned long long ga = (unsigned long long)(size_t)
                (VT + ((size_t)b * D + w * 64) * L + jt);
            const unsigned ldsA =
                (unsigned)(size_t)(&Vt[0]) + (unsigned)(w * 64 * 64); // bytes

            u32x4 g0;
            g0[0] = 1u;                                     // count=1
            g0[1] = ldsA;                                   // lds_addr
            g0[2] = (unsigned)ga;                           // global_addr lo
            g0[3] = ((unsigned)(ga >> 32) & 0x01FFFFFFu) | (2u << 30); // type=2

            i32x8 g1;
            g1[0] = (int)(1u << 16);                        // data_size=2B
            g1[1] = (int)(((unsigned)L & 0xFFFFu) << 16);   // tensor_dim0 lo16
            g1[2] = (int)((((unsigned)L >> 16) & 0xFFFFu) | (64u << 16)); // dim0 hi | dim1 lo
            g1[3] = (int)(32u << 16);                       // tile_dim0 = 32
            g1[4] = 64;                                     // tile_dim1 = 64
            g1[5] = L;                                      // dim0_stride lo32
            g1[6] = 0;                                      // dim0_stride hi | dim1_stride lo
            g1[7] = 0;

            i32x4 g2 = (i32x4)0, g3 = (i32x4)0;
#if __clang_major__ >= 23
            __builtin_amdgcn_tensor_load_to_lds(g0, g1, g2, g3, (i32x8)0, 0);
#else
            __builtin_amdgcn_tensor_load_to_lds(g0, g1, g2, g3, 0);
#endif
            __builtin_amdgcn_s_wait_tensorcnt(0);
        }
#else
        for (int r = tid; r < 512; r += 256) {
            const _Float16* s = VT + ((size_t)b * D + r) * L + jt;
            _Float16* dls = &Vt[r * 32];
            #pragma unroll
            for (int u = 0; u < 32; ++u) dls[u] = s[u];
        }
#endif
        __syncthreads();

        v16h a[2], bf[4];
        #pragma unroll
        for (int mi = 0; mi < 2; ++mi)
            a[mi] = *(const v16h*)&Pls[(mi * 16 + lr) * 32 + khv * 16];
        #pragma unroll
        for (int ni = 0; ni < 4; ++ni)
            bf[ni] = *(const v16h*)&Vt[(d0 + ni * 16 + lr) * 32 + khv * 16];

        #pragma unroll
        for (int mi = 0; mi < 2; ++mi)
            #pragma unroll
            for (int ni = 0; ni < 4; ++ni)
                acc[mi][ni] = __builtin_amdgcn_wmma_f32_16x16x32_f16(
                    false, a[mi], false, bf[ni], (short)0, acc[mi][ni],
                    false, false);
        __syncthreads();
    }

    // Epilogue: scale by 1/rowsum, store.
    const int crow = (lane >> 4) * 8;
    const int ccol = lane & 15;
    #pragma unroll
    for (int mi = 0; mi < 2; ++mi) {
        #pragma unroll
        for (int ni = 0; ni < 4; ++ni) {
            #pragma unroll
            for (int v = 0; v < 8; ++v) {
                const int li = mi * 16 + crow + v;
                const int gd = d0 + ni * 16 + ccol;
                out[(tokBase + i0 + li) * D + gd] = acc[mi][ni][v] * rinvLds[li];
            }
        }
    }
}

// ---------------------------------------------------------------------------
// Segmented cumsum over the query axis (L=2048, 8 segments of 256).
// ---------------------------------------------------------------------------
__global__ __launch_bounds__(256) void cumsum_part(
    float* __restrict__ X, float* __restrict__ aux, int L, int D)
{
    const int b = blockIdx.z, s = blockIdx.y;
    const int d = blockIdx.x * 256 + threadIdx.x;
    float* p = X + ((size_t)b * L + s * 256) * D + d;
    float run = 0.0f;
    for (int i = 0; i < 256; ++i) {
        run += p[(size_t)i * D];
        p[(size_t)i * D] = run;
    }
    aux[((size_t)b * 8 + s) * D + d] = run;
}

__global__ __launch_bounds__(256) void cumsum_fix(
    float* __restrict__ X, const float* __restrict__ aux, int L, int D)
{
    const int b = blockIdx.z, s = blockIdx.y;
    const int d = blockIdx.x * 256 + threadIdx.x;
    if (s == 0) return;
    float off = 0.0f;
    for (int t = 0; t < s; ++t) off += aux[((size_t)b * 8 + t) * D + d];
    float* p = X + ((size_t)b * L + s * 256) * D + d;
    for (int i = 0; i < 256; ++i) p[(size_t)i * D] += off;
}

// ---------------------------------------------------------------------------
extern "C" void kernel_launch(void* const* d_in, const int* in_sizes, int n_in,
                              void* d_out, int out_size, void* d_ws, size_t ws_size,
                              hipStream_t stream)
{
    (void)in_sizes; (void)n_in; (void)out_size; (void)ws_size;
    const float* query = (const float*)d_in[0];
    const float* key_  = (const float*)d_in[1];
    const float* value = (const float*)d_in[2];
    const float* wq_w  = (const float*)d_in[3];
    const float* wq_b  = (const float*)d_in[4];
    const float* wk_w  = (const float*)d_in[5];
    const float* wk_b  = (const float*)d_in[6];
    const float* wv_w  = (const float*)d_in[7];
    const float* wv_b  = (const float*)d_in[8];
    const float* fc_w  = (const float*)d_in[9];
    const float* fc_b  = (const float*)d_in[10];

    const int B = 8, L = 2048, D = 512;
    const int M = B * L;  // 16384

    char* ws = (char*)d_ws;
    float*    Qf  = (float*)(ws + 0);
    float*    Kf  = (float*)(ws + (size_t)33554432);
    float*    AO  = (float*)(ws + (size_t)67108864);
    _Float16* Vh  = (_Float16*)(ws + (size_t)100663296);
    _Float16* VTg = (_Float16*)(ws + (size_t)117440512);
    float*    st  = (float*)(ws + (size_t)134217728);
    float* qm = st,         * qz = st + 16384, * qs = st + 32768, * qh  = st + 49152;
    float* km = st + 65536, * kz = st + 81920, * ks = st + 98304, * khs = st + 114688;
    float*    aux = (float*)(ws + (size_t)134742016);

    const dim3 ggrid(D / BN, M / BM);   // (8, 128)

    gemm_xwT<false><<<ggrid, 256, 0, stream>>>(query, wq_w, wq_b, Qf, M, D, D);
    gemm_xwT<false><<<ggrid, 256, 0, stream>>>(key_,  wk_w, wk_b, Kf, M, D, D);
    gemm_xwT<true ><<<ggrid, 256, 0, stream>>>(value, wv_w, wv_b, Vh, M, D, D);

    token_stats<<<M / 8, 256, 0, stream>>>(Qf, qm, qz, qs, qh, D);
    token_stats<<<M / 8, 256, 0, stream>>>(Kf, km, kz, ks, khs, D);

    transpose_v<<<dim3(L / 64, D / 64, B), 256, 0, stream>>>(Vh, VTg, L, D);

    attn_pv<<<dim3(L / 32, B), 256, 0, stream>>>(qm, qz, qs, qh,
                                                 km, kz, ks, khs,
                                                 VTg, AO, L, D);

    cumsum_part<<<dim3(D / 256, 8, B), 256, 0, stream>>>(AO, aux, L, D);
    cumsum_fix <<<dim3(D / 256, 8, B), 256, 0, stream>>>(AO, aux, L, D);

    gemm_xwT<false><<<ggrid, 256, 0, stream>>>(AO, fc_w, fc_b, (float*)d_out, M, D, D);
}
